// Model_default_25769804009
// MI455X (gfx1250) — matrix operand order repaired and checked by reading, not compiled, not run on hardware
//
#include <hip/hip_runtime.h>
#include <cmath>

// ---------------------------------------------------------------------------
// CDNA5 (gfx1250) WMMA types and helpers
// ---------------------------------------------------------------------------
typedef __attribute__((ext_vector_type(16))) _Float16 v16h;
typedef __attribute__((ext_vector_type(8)))  float    v8f;
typedef unsigned int v4u __attribute__((ext_vector_type(4)));
typedef int          v4i __attribute__((ext_vector_type(4)));
typedef int          v8i __attribute__((ext_vector_type(8)));

__device__ __forceinline__ v8f wmma_f32_f16(v16h a, v16h b, v8f c) {
  // D = A(16x32 f16) x B(32x16 f16) + C(16x16 f32)
  return __builtin_amdgcn_wmma_f32_16x16x32_f16(
      /*neg_a=*/false, a, /*neg_b=*/false, b,
      /*c_mod=*/(short)0, c, /*reuse_a=*/false, /*reuse_b=*/false);
}

// Fast activations: one v_exp_f32 + one v_rcp_f32 each (inputs are bounded).
__device__ __forceinline__ float sigm(float x) {
  return __builtin_amdgcn_rcpf(1.f + __expf(-x));
}
__device__ __forceinline__ float tanh_fast(float x) {
  float e2 = __expf(2.f * x);                       // inf/0 saturate correctly
  return 1.f - 2.f * __builtin_amdgcn_rcpf(e2 + 1.f);
}

// GROUPS = [{2}, {4..25}, {26..32}, {34,36}, {33}, {37}]
__device__ __forceinline__ int group_of(int t) {
  if (t == 2) return 0;
  if (t >= 4 && t <= 25) return 1;
  if (t >= 26 && t <= 32) return 2;
  if (t == 34 || t == 36) return 3;
  if (t == 33) return 4;
  if (t == 37) return 5;
  return -1;
}

// A-fragment element e -> K index (within a K=32 chunk), contiguous order:
//   lanes 0-15 : e0..15 -> K0..15
//   lanes16-31 : e0..15 -> K16..31
#define AFRAG_K(half_, e_) ((half_) * 16 + (e_))

// ---------------------------------------------------------------------------
// Tensor Data Mover: 1-D tile copy global(f16) -> LDS.  D# per ISA ch.8:
//  group0: count=1, lds_addr, global_addr[56:0], type=2
//  group1: data_size=1(2B), tensor_dim0=tile_dim0=nelem, dim0_stride=nelem
// ---------------------------------------------------------------------------
#if __has_builtin(__builtin_amdgcn_tensor_load_to_lds)
#define HAVE_TDM 1
__device__ __forceinline__ void tdm_load_f16(_Float16* lds_dst,
                                             const _Float16* gsrc, int nelem) {
  unsigned long long ga = (unsigned long long)(const void*)gsrc;
  unsigned int lo = (unsigned int)(unsigned long long)(void*)lds_dst;
  v4u g0;
  g0[0] = 1u;                                            // count=1 (valid user D#)
  g0[1] = lo;                                            // lds_addr (bytes)
  g0[2] = (unsigned int)(ga & 0xFFFFFFFFu);              // global_addr[31:0]
  g0[3] = (unsigned int)((ga >> 32) & 0x01FFFFFFu)       // global_addr[56:32]
          | 0x80000000u;                                 // type=2 ("image")
  v8i g1;
  g1[0] = 0x00010000;                                    // data_size=1 (2 bytes)
  g1[1] = (int)((unsigned)(nelem & 0xFFFF) << 16);       // tensor_dim0[15:0]
  g1[2] = (int)(((unsigned)nelem >> 16) | (1u << 16));   // dim0[31:16], tensor_dim1=1
  g1[3] = (int)((unsigned)(nelem & 0xFFFF) << 16);       // tile_dim0 = nelem
  g1[4] = 0;                                             // tile_dim1/2 unused
  g1[5] = nelem;                                         // tensor_dim0_stride lo
  g1[6] = 0;
  g1[7] = 0;
  v4i gz = {0, 0, 0, 0};
#if defined(__clang_major__) && (__clang_major__ >= 23)
  v8i gz8 = {0, 0, 0, 0, 0, 0, 0, 0};
  __builtin_amdgcn_tensor_load_to_lds(g0, g1, gz, gz, gz8, 0);
#else
  __builtin_amdgcn_tensor_load_to_lds(g0, g1, gz, gz, 0);
#endif
}
#else
#define HAVE_TDM 0
#endif

// ---------------------------------------------------------------------------
// Misc prep kernels
// ---------------------------------------------------------------------------
__global__ void zero_init_kernel(float* aggr, int n, int* meta) {
  int i = blockIdx.x * blockDim.x + threadIdx.x;
  if (i < 32) meta[i] = 0;
  if (i < n) aggr[i] = 0.f;
}

__global__ void cvt_f16_kernel(const float* __restrict__ s, _Float16* __restrict__ d, int n) {
  int i = blockIdx.x * blockDim.x + threadIdx.x;
  if (i < n) d[i] = (_Float16)s[i];
}

__global__ void pe_kernel(float* __restrict__ pe) {
  int i = blockIdx.x * blockDim.x + threadIdx.x;
  if (i >= 8 * 128) return;
  int p = i >> 7, c = i & 127;
  float f = __expf((float)(c & ~1) * (-9.210340371976184f / 128.f)); // -ln(10000)/128
  float ang = (float)p * f;
  pe[i] = (c & 1) ? __cosf(ang) : __sinf(ang);
}

__global__ void count_kernel(const int* __restrict__ node_type,
                             const int* __restrict__ edge_index, int* meta, int E) {
  int e = blockIdx.x * blockDim.x + threadIdx.x;
  if (e >= E) return;
  int g = group_of(node_type[edge_index[E + e]]);
  if (g >= 0) atomicAdd(&meta[g], 1);
}

__global__ void scan_kernel(int* meta) {
  if (threadIdx.x == 0 && blockIdx.x == 0) {
    int acc = 0;
    for (int g = 0; g < 6; ++g) { meta[8 + g] = acc; acc += meta[g]; meta[16 + g] = 0; }
    meta[14] = acc;
  }
}

__global__ void scatter_kernel(const int* __restrict__ node_type,
                               const int* __restrict__ edge_index,
                               int* meta, int* bucket, int E) {
  int e = blockIdx.x * blockDim.x + threadIdx.x;
  if (e >= E) return;
  int g = group_of(node_type[edge_index[E + e]]);
  if (g < 0) return;
  int pos = meta[8 + g] + atomicAdd(&meta[16 + g], 1);
  bucket[pos] = e;
}

// ---------------------------------------------------------------------------
// 2-layer GRU encoder: 4 waves/block, 16 nodes per wave, h-state in LDS.
// ---------------------------------------------------------------------------
__global__ __launch_bounds__(128) void encoder_kernel(
    const float* __restrict__ sim_res, const int* __restrict__ node_type,
    const _Float16* __restrict__ Wih0, const _Float16* __restrict__ Whh0,
    const _Float16* __restrict__ Wih1, const _Float16* __restrict__ Whh1,
    const float* __restrict__ bih0, const float* __restrict__ bhh0,
    const float* __restrict__ bih1, const float* __restrict__ bhh1,
    float* __restrict__ node_emb, int N) {
  __shared__ float shh[4][2][16][64];  // per-wave h1 / h2, 32 KB
  const int w = threadIdx.x >> 5;
  const int lane = threadIdx.x & 31;
  const int half = lane >> 4;
  const int m = lane & 15;
  const int nb = blockIdx.x * 64 + w * 16;
  const int node = nb + m;
  const bool nv = node < N;

  float* h1 = &shh[w][0][0][0];
  float* h2 = &shh[w][1][0][0];
  for (int i = lane; i < 2 * 16 * 64; i += 32) (&shh[w][0][0][0])[i] = 0.f;

  const float* xbase = sim_res + (size_t)(nv ? node : 0) * 32 * 32;
  __builtin_prefetch(xbase, 0, 0);  // global_prefetch_b8 on the input stream

  for (int t = 0; t < 32; ++t) {
    // ---------------- layer 0 : x(K=32) , h1(K=64) ----------------
    v16h ax;
    {
      const float* xr = xbase + t * 32;
#pragma unroll
      for (int e = 0; e < 16; ++e) {
        int k = AFRAG_K(half, e);
        ax[e] = nv ? (_Float16)xr[k] : (_Float16)0.f;
      }
    }
    v16h ah[2];
#pragma unroll
    for (int kc = 0; kc < 2; ++kc)
#pragma unroll
      for (int e = 0; e < 16; ++e)
        ah[kc][e] = (_Float16)h1[m * 64 + kc * 32 + AFRAG_K(half, e)];

    v8f hn0[4];
    for (int j = 0; j < 4; ++j) {
      v8f gi[3], gh[3];
#pragma unroll
      for (int gate = 0; gate < 3; ++gate) {
        const int rowb = gate * 64 + j * 16;
        v8f ci, ch;
        const float bi = bih0[rowb + m];
        const float bh = bhh0[rowb + m];
#pragma unroll
        for (int r = 0; r < 8; ++r) { ci[r] = bi; ch[r] = bh; }
        v16h bx = *(const v16h*)(Wih0 + (rowb + m) * 32 + half * 16);
        ci = wmma_f32_f16(ax, bx, ci);
        v16h bh0 = *(const v16h*)(Whh0 + (rowb + m) * 64 + half * 16);
        v16h bh1 = *(const v16h*)(Whh0 + (rowb + m) * 64 + 32 + half * 16);
        ch = wmma_f32_f16(ah[0], bh0, ch);
        ch = wmma_f32_f16(ah[1], bh1, ch);
        gi[gate] = ci; gh[gate] = ch;
      }
#pragma unroll
      for (int r = 0; r < 8; ++r) {
        float hold = h1[(r + 8 * half) * 64 + j * 16 + m];
        float rg = sigm(gi[0][r] + gh[0][r]);
        float zg = sigm(gi[1][r] + gh[1][r]);
        float ng = tanh_fast(gi[2][r] + rg * gh[2][r]);
        hn0[j][r] = (1.f - zg) * ng + zg * hold;
      }
    }
#pragma unroll
    for (int j = 0; j < 4; ++j)
#pragma unroll
      for (int r = 0; r < 8; ++r)
        h1[(r + 8 * half) * 64 + j * 16 + m] = hn0[j][r];

    // ---------------- layer 1 : x = new h1 (K=64), h2 (K=64) ----------------
    v16h a1[2], a2h[2];
#pragma unroll
    for (int kc = 0; kc < 2; ++kc)
#pragma unroll
      for (int e = 0; e < 16; ++e) {
        int k = kc * 32 + AFRAG_K(half, e);
        a1[kc][e]  = (_Float16)h1[m * 64 + k];
        a2h[kc][e] = (_Float16)h2[m * 64 + k];
      }
    v8f hn1[4];
    for (int j = 0; j < 4; ++j) {
      v8f gi[3], gh[3];
#pragma unroll
      for (int gate = 0; gate < 3; ++gate) {
        const int rowb = gate * 64 + j * 16;
        v8f ci, ch;
        const float bi = bih1[rowb + m];
        const float bh = bhh1[rowb + m];
#pragma unroll
        for (int r = 0; r < 8; ++r) { ci[r] = bi; ch[r] = bh; }
        v16h bi0 = *(const v16h*)(Wih1 + (rowb + m) * 64 + half * 16);
        v16h bi1 = *(const v16h*)(Wih1 + (rowb + m) * 64 + 32 + half * 16);
        ci = wmma_f32_f16(a1[0], bi0, ci);
        ci = wmma_f32_f16(a1[1], bi1, ci);
        v16h bq0 = *(const v16h*)(Whh1 + (rowb + m) * 64 + half * 16);
        v16h bq1 = *(const v16h*)(Whh1 + (rowb + m) * 64 + 32 + half * 16);
        ch = wmma_f32_f16(a2h[0], bq0, ch);
        ch = wmma_f32_f16(a2h[1], bq1, ch);
        gi[gate] = ci; gh[gate] = ch;
      }
#pragma unroll
      for (int r = 0; r < 8; ++r) {
        float hold = h2[(r + 8 * half) * 64 + j * 16 + m];
        float rg = sigm(gi[0][r] + gh[0][r]);
        float zg = sigm(gi[1][r] + gh[1][r]);
        float ng = tanh_fast(gi[2][r] + rg * gh[2][r]);
        hn1[j][r] = (1.f - zg) * ng + zg * hold;
      }
    }
#pragma unroll
    for (int j = 0; j < 4; ++j)
#pragma unroll
      for (int r = 0; r < 8; ++r)
        h2[(r + 8 * half) * 64 + j * 16 + m] = hn1[j][r];
  }

  // node_emb = io_mask ? concat(h1, h2) : 0
  for (int i = lane; i < 16 * 128; i += 32) {
    int mm = i >> 7, c = i & 127;
    int nd = nb + mm;
    if (nd < N) {
      bool io = node_type[nd] <= 1;  // INPUT_IDX or CONST_IDX
      float v = io ? (c < 64 ? h1[mm * 64 + c] : h2[mm * 64 + (c - 64)]) : 0.f;
      node_emb[(size_t)nd * 128 + c] = v;
    }
  }
}

// ---------------------------------------------------------------------------
// Message passing: grid.y = group, each wave = 16 same-group edges.
// Per-group W1/W2 staged to LDS via TDM (tensor_load_to_lds + s_wait_tensorcnt).
// msg = relu((x_src+pe+op_emb) W1^T + b1) W2^T + b2 ; atomicAdd into aggr[dst]
// ---------------------------------------------------------------------------
__global__ __launch_bounds__(256) void mp_kernel(
    const int* __restrict__ node_type, const int* __restrict__ edge_index,
    const int* __restrict__ edge_type, const float* __restrict__ node_emb,
    const float* __restrict__ pe, const float* __restrict__ op_emb,
    const _Float16* __restrict__ W1, const _Float16* __restrict__ W2,
    const float* __restrict__ b1, const float* __restrict__ b2,
    const int* __restrict__ bucket, const int* __restrict__ meta,
    float* __restrict__ aggr, int E) {
  __shared__ __align__(128) _Float16 sW1[128 * 128];   // 32 KB
  __shared__ __align__(128) _Float16 sW2[128 * 128];   // 32 KB
  __shared__ int sdst[8][16];
  __shared__ _Float16 smid[8][16][128];                // 32 KB

  const int g = blockIdx.y;
  const int o0 = meta[8 + g];
  const int cnt = meta[9 + g] - o0;
  const int w = threadIdx.x >> 5;
  const int lane = threadIdx.x & 31;
  const int half = lane >> 4;
  const int m = lane & 15;
  const int tb = blockIdx.x * 128 + w * 16;
  const bool active = tb < cnt;

  const _Float16* W1g = W1 + (size_t)g * 128 * 128;
  const _Float16* W2g = W2 + (size_t)g * 128 * 128;
#if HAVE_TDM
  if (w == 0) {  // one wave drives the TDM (EXEC ignored by tensor ops)
    tdm_load_f16(sW1, W1g, 128 * 128);
    tdm_load_f16(sW2, W2g, 128 * 128);
#if __has_builtin(__builtin_amdgcn_s_wait_tensorcnt)
    __builtin_amdgcn_s_wait_tensorcnt(0);
#endif
  }
#else
  for (int i = threadIdx.x; i < 128 * 128; i += 256) {
    sW1[i] = W1g[i];
    sW2[i] = W2g[i];
  }
#endif
  __syncthreads();
  if (!active) return;  // wave-uniform; EXEC stays full for WMMA below

  const int slot = tb + m;
  const bool ev = slot < cnt;
  const int eid = ev ? bucket[o0 + slot] : 0;
  const int srcn = edge_index[eid];
  const int dstn = edge_index[E + eid];
  const int et = edge_type[eid];
  const int op = node_type[dstn];
  if (half == 0) sdst[w][m] = ev ? dstn : -1;

  v16h a[4];
  {
    const float* xs = node_emb + (size_t)srcn * 128;
    const float* pp = pe + et * 128;
    const float* oo = op_emb + ((size_t)g * 40 + op) * 128;
    __builtin_prefetch(xs, 0, 3);
#pragma unroll
    for (int kc = 0; kc < 4; ++kc)
#pragma unroll
      for (int e = 0; e < 16; ++e) {
        int k = kc * 32 + AFRAG_K(half, e);
        a[kc][e] = ev ? (_Float16)(xs[k] + pp[k] + oo[k]) : (_Float16)0.f;
      }
  }
  for (int j = 0; j < 8; ++j) {
    v8f c;
    const float bb = b1[g * 128 + j * 16 + m];
#pragma unroll
    for (int r = 0; r < 8; ++r) c[r] = bb;
#pragma unroll
    for (int kc = 0; kc < 4; ++kc) {
      v16h b = *(const v16h*)(sW1 + (j * 16 + m) * 128 + kc * 32 + half * 16);
      c = wmma_f32_f16(a[kc], b, c);
    }
#pragma unroll
    for (int r = 0; r < 8; ++r)
      smid[w][r + 8 * half][j * 16 + m] = (_Float16)fmaxf(c[r], 0.f);
  }
  v16h a2[4];
#pragma unroll
  for (int kc = 0; kc < 4; ++kc)
#pragma unroll
    for (int e = 0; e < 16; ++e)
      a2[kc][e] = smid[w][m][kc * 32 + AFRAG_K(half, e)];
  for (int j = 0; j < 8; ++j) {
    v8f c;
    const float bb = b2[g * 128 + j * 16 + m];
#pragma unroll
    for (int r = 0; r < 8; ++r) c[r] = bb;
#pragma unroll
    for (int kc = 0; kc < 4; ++kc) {
      v16h b = *(const v16h*)(sW2 + (j * 16 + m) * 128 + kc * 32 + half * 16);
      c = wmma_f32_f16(a2[kc], b, c);
    }
#pragma unroll
    for (int r = 0; r < 8; ++r) {
      int d = sdst[w][r + 8 * half];
      if (d >= 0) atomicAdd(&aggr[(size_t)d * 128 + j * 16 + m], c[r]);
    }
  }
}

// ---------------------------------------------------------------------------
// Single-step GRU update: out = gru_cell(aggr, node_emb)
// ---------------------------------------------------------------------------
__global__ __launch_bounds__(256) void update_kernel(
    const float* __restrict__ aggr, const float* __restrict__ node_emb,
    const _Float16* __restrict__ Wih, const _Float16* __restrict__ Whh,
    const float* __restrict__ bih, const float* __restrict__ bhh,
    float* __restrict__ out, int N) {
  const int w = threadIdx.x >> 5;
  const int lane = threadIdx.x & 31;
  const int half = lane >> 4;
  const int m = lane & 15;
  const int nb = blockIdx.x * 128 + w * 16;
  const int node = nb + m;
  const bool nv = node < N;

  v16h ax[4], ah[4];
  {
    const float* xr = aggr + (size_t)(nv ? node : 0) * 128;
    const float* hr = node_emb + (size_t)(nv ? node : 0) * 128;
    __builtin_prefetch(hr, 0, 3);
#pragma unroll
    for (int kc = 0; kc < 4; ++kc)
#pragma unroll
      for (int e = 0; e < 16; ++e) {
        int k = kc * 32 + AFRAG_K(half, e);
        ax[kc][e] = nv ? (_Float16)xr[k] : (_Float16)0.f;
        ah[kc][e] = nv ? (_Float16)hr[k] : (_Float16)0.f;
      }
  }
  for (int j = 0; j < 8; ++j) {
    v8f gi[3], gh[3];
#pragma unroll
    for (int gate = 0; gate < 3; ++gate) {
      const int rowb = gate * 128 + j * 16;
      v8f ci, ch;
      const float bi = bih[rowb + m];
      const float bh = bhh[rowb + m];
#pragma unroll
      for (int r = 0; r < 8; ++r) { ci[r] = bi; ch[r] = bh; }
#pragma unroll
      for (int kc = 0; kc < 4; ++kc) {
        v16h bwi = *(const v16h*)(Wih + (size_t)(rowb + m) * 128 + kc * 32 + half * 16);
        ci = wmma_f32_f16(ax[kc], bwi, ci);
        v16h bwh = *(const v16h*)(Whh + (size_t)(rowb + m) * 128 + kc * 32 + half * 16);
        ch = wmma_f32_f16(ah[kc], bwh, ch);
      }
      gi[gate] = ci; gh[gate] = ch;
    }
#pragma unroll
    for (int r = 0; r < 8; ++r) {
      int nd = nb + r + 8 * half;
      bool v = nd < N;
      float hold = v ? node_emb[(size_t)nd * 128 + j * 16 + m] : 0.f;
      float rg = sigm(gi[0][r] + gh[0][r]);
      float zg = sigm(gi[1][r] + gh[1][r]);
      float ng = tanh_fast(gi[2][r] + rg * gh[2][r]);
      if (v) out[(size_t)nd * 128 + j * 16 + m] = (1.f - zg) * ng + zg * hold;
    }
  }
}

// ---------------------------------------------------------------------------
// Host entry
// ---------------------------------------------------------------------------
extern "C" void kernel_launch(void* const* d_in, const int* in_sizes, int n_in,
                              void* d_out, int out_size, void* d_ws, size_t ws_size,
                              hipStream_t stream) {
  (void)n_in; (void)out_size; (void)ws_size;
  const int N = in_sizes[0];
  const int E = in_sizes[2];
  const int* node_type   = (const int*)d_in[0];
  const int* edge_index  = (const int*)d_in[1];
  const int* edge_type   = (const int*)d_in[2];
  const float* sim_res   = (const float*)d_in[3];
  const float* enc_Wih0  = (const float*)d_in[4];
  const float* enc_Whh0  = (const float*)d_in[5];
  const float* enc_bih0  = (const float*)d_in[6];
  const float* enc_bhh0  = (const float*)d_in[7];
  const float* enc_Wih1  = (const float*)d_in[8];
  const float* enc_Whh1  = (const float*)d_in[9];
  const float* enc_bih1  = (const float*)d_in[10];
  const float* enc_bhh1  = (const float*)d_in[11];
  const float* upd_Wih   = (const float*)d_in[12];
  const float* upd_Whh   = (const float*)d_in[13];
  const float* upd_bih   = (const float*)d_in[14];
  const float* upd_bhh   = (const float*)d_in[15];
  const float* grp_W1    = (const float*)d_in[16];
  const float* grp_b1    = (const float*)d_in[17];
  const float* grp_W2    = (const float*)d_in[18];
  const float* grp_b2    = (const float*)d_in[19];
  const float* grp_opemb = (const float*)d_in[20];

  char* ws = (char*)d_ws;
  size_t off = 0;
  auto alloc = [&](size_t bytes) -> void* {
    off = (off + 255) & ~(size_t)255;
    void* p = ws + off;
    off += bytes;
    return p;
  };
  float* node_emb = (float*)alloc((size_t)N * 128 * 4);
  float* aggr     = (float*)alloc((size_t)N * 128 * 4);
  int*   bucket   = (int*)alloc((size_t)E * 4);
  int*   meta     = (int*)alloc(32 * 4);
  float* pe       = (float*)alloc(8 * 128 * 4);
  _Float16* hWih0 = (_Float16*)alloc(192 * 32 * 2);
  _Float16* hWhh0 = (_Float16*)alloc(192 * 64 * 2);
  _Float16* hWih1 = (_Float16*)alloc(192 * 64 * 2);
  _Float16* hWhh1 = (_Float16*)alloc(192 * 64 * 2);
  _Float16* hUWih = (_Float16*)alloc(384 * 128 * 2);
  _Float16* hUWhh = (_Float16*)alloc(384 * 128 * 2);
  _Float16* hW1   = (_Float16*)alloc((size_t)6 * 128 * 128 * 2);
  _Float16* hW2   = (_Float16*)alloc((size_t)6 * 128 * 128 * 2);

  const int nAggr = N * 128;
  zero_init_kernel<<<(nAggr + 255) / 256, 256, 0, stream>>>(aggr, nAggr, meta);
  auto cvt = [&](const float* s, _Float16* d, int n) {
    cvt_f16_kernel<<<(n + 255) / 256, 256, 0, stream>>>(s, d, n);
  };
  cvt(enc_Wih0, hWih0, 192 * 32);
  cvt(enc_Whh0, hWhh0, 192 * 64);
  cvt(enc_Wih1, hWih1, 192 * 64);
  cvt(enc_Whh1, hWhh1, 192 * 64);
  cvt(upd_Wih, hUWih, 384 * 128);
  cvt(upd_Whh, hUWhh, 384 * 128);
  cvt(grp_W1, hW1, 6 * 128 * 128);
  cvt(grp_W2, hW2, 6 * 128 * 128);
  pe_kernel<<<4, 256, 0, stream>>>(pe);

  encoder_kernel<<<(N + 63) / 64, 128, 0, stream>>>(
      sim_res, node_type, hWih0, hWhh0, hWih1, hWhh1,
      enc_bih0, enc_bhh0, enc_bih1, enc_bhh1, node_emb, N);

  count_kernel<<<(E + 255) / 256, 256, 0, stream>>>(node_type, edge_index, meta, E);
  scan_kernel<<<1, 32, 0, stream>>>(meta);
  scatter_kernel<<<(E + 255) / 256, 256, 0, stream>>>(node_type, edge_index, meta, bucket, E);

  dim3 mpg((E + 127) / 128, 6);
  mp_kernel<<<mpg, 256, 0, stream>>>(node_type, edge_index, edge_type, node_emb,
                                     pe, grp_opemb, hW1, hW2, grp_b1, grp_b2,
                                     bucket, meta, aggr, E);

  update_kernel<<<(N + 127) / 128, 256, 0, stream>>>(aggr, node_emb, hUWih, hUWhh,
                                                     upd_bih, upd_bhh, (float*)d_out, N);
}
